// BilinearAttention_2723009266063
// MI455X (gfx1250) — matrix operand order repaired and checked by reading, not compiled, not run on hardware
//
#include <hip/hip_runtime.h>
#include <hip/hip_bf16.h>
#include <cstddef>

// ---------------------------------------------------------------------------
// BilinearAttention for MI455X (gfx1250): bf16 WMMA (v_wmma_f32_16x16x32_bf16)
// for all four matmul stages, RoPE fused in registers, causal mask from
// indices, LDS relayout (ds) for the score->A-matrix transpose.
// ---------------------------------------------------------------------------

#define D_MODEL 1024
#define D_HEAD  128
#define BATCH   4
#define SEQ     4096
#define ROWS    (BATCH * SEQ)   // 16384 flat (b,s) rows

typedef __bf16 bf16_t;
typedef __attribute__((ext_vector_type(16))) __bf16        v16bf;
typedef __attribute__((ext_vector_type(8)))  float         v8f;
typedef __attribute__((ext_vector_type(8)))  unsigned short u16x8;

union v16bf_cast { v16bf v; u16x8 u[2]; };

__device__ __forceinline__ unsigned short f2bs(float x) {
    __bf16 b = (__bf16)x;
    return __builtin_bit_cast(unsigned short, b);
}

// WMMA D = A(16x32 bf16) * B(32x16 bf16) + C(16x16 f32)
__device__ __forceinline__ v8f wmma_bf16(v16bf a, v16bf b, v8f c) {
    return __builtin_amdgcn_wmma_f32_16x16x32_bf16(
        /*neg_a=*/false, a, /*neg_b=*/false, b,
        /*c_mod=*/(short)0, c, /*reuse_a=*/false, /*reuse_b=*/false);
}

// A-matrix 16x32 bf16 tile from row-major [16 x >=32] storage.
// Lane l: row = l%16, h = l/16; elements 0..7 at K = h*8.., 8..15 at K = 16+h*8..
__device__ __forceinline__ v16bf load_tile_A(const unsigned short* base,
                                             int stride, int lane) {
    int r = lane & 15, h = lane >> 4;
    v16bf_cast t;
    const unsigned short* p = base + (size_t)r * stride;
    t.u[0] = *(const u16x8*)(p + h * 8);
    t.u[1] = *(const u16x8*)(p + 16 + h * 8);
    return t.v;
}

// B-matrix 32x16 bf16 tile; source stored as [N rows x K cols] row-major
// (i.e. B[k][n] = src[n][k]).  Lane l: n = l%16, elements 0..7 at K = h*8.., 8..15 at K = 16+h*8..
__device__ __forceinline__ v16bf load_tile_B(const unsigned short* base,
                                             int stride, int lane) {
    int n = lane & 15, h = lane >> 4;
    v16bf_cast t;
    const unsigned short* p = base + (size_t)n * stride;
    t.u[0] = *(const u16x8*)(p + h * 8);
    t.u[1] = *(const u16x8*)(p + 16 + h * 8);
    return t.v;
}

// A-matrix tile sourced from fp32 row-major data, converted to bf16 on the fly.
__device__ __forceinline__ v16bf load_tile_A_f32(const float* base,
                                                 int stride, int lane) {
    int r = lane & 15, h = lane >> 4;
    const float* p0 = base + (size_t)r * stride + h * 8;
    const float* p1 = base + (size_t)r * stride + 16 + h * 8;
    v16bf v;
#pragma unroll
    for (int i = 0; i < 8; ++i) {
        v[i]     = (__bf16)p0[i];
        v[8 + i] = (__bf16)p1[i];
    }
    return v;
}

// ---------------------------------------------------------------------------
// Kernel 1: convert + transpose weights into bf16 WMMA-friendly layouts.
//   Wcat (640 x 1024): row o = head*128+d, contiguous along D  (B-tile source)
//   Wot  (1024 x 128): row Do, contiguous along d              (B-tile source)
// ---------------------------------------------------------------------------
__global__ void prep_weights(const float* __restrict__ Wq1,
                             const float* __restrict__ Wq2,
                             const float* __restrict__ Wk1,
                             const float* __restrict__ Wk2,
                             const float* __restrict__ Wv,
                             const float* __restrict__ Wo,
                             unsigned short* __restrict__ Wcat,
                             unsigned short* __restrict__ Wot) {
    int idx = blockIdx.x * blockDim.x + threadIdx.x;
    if (idx < 640 * 1024) {
        int o = idx >> 10;          // output channel 0..639
        int D = idx & 1023;
        int head = o >> 7;
        int d = o & 127;
        const float* Ws[5] = {Wq1, Wq2, Wk1, Wk2, Wv};
        Wcat[idx] = f2bs(Ws[head][(size_t)D * D_HEAD + d]);
    } else {
        int j = idx - 640 * 1024;
        if (j < 1024 * 128) {
            int Do = j >> 7;
            int d = j & 127;
            Wot[j] = f2bs(Wo[(size_t)d * D_MODEL + Do]);
        }
    }
}

// ---------------------------------------------------------------------------
// Kernel 2: fused projection + RoPE.  One wave = 16 rows x one full head
// (8 accumulator tiles, K=1024 -> 32 chunks -> 256 WMMAs/wave).
// q1,q2,k1,k2 -> bf16 row-major (ROWS x 128) with RoPE applied.
// v -> bf16 transposed Vt[b][d][s] so A*V B-tiles are contiguous along t.
// ---------------------------------------------------------------------------
__global__ void __launch_bounds__(128)
proj_rope(const float* __restrict__ x,
          const float* __restrict__ cosT, const float* __restrict__ sinT,
          const unsigned short* __restrict__ Wcat,
          unsigned short* __restrict__ q1w, unsigned short* __restrict__ q2w,
          unsigned short* __restrict__ k1w, unsigned short* __restrict__ k2w,
          unsigned short* __restrict__ Vt) {
    int lane = threadIdx.x & 31;
    int gw = blockIdx.x * 4 + (threadIdx.x >> 5);   // 5120 waves
    int head = gw % 5;
    int rb = gw / 5;                                 // 0..1023
    int s0 = rb * 16;                                // flat (b,s) row base

    v8f acc[8] = {};
#pragma unroll 1
    for (int kc = 0; kc < D_MODEL; kc += 32) {
        v16bf a = load_tile_A_f32(x + (size_t)s0 * D_MODEL + kc, D_MODEL, lane);
#pragma unroll
        for (int i = 0; i < 8; ++i) {
            v16bf b = load_tile_B(
                Wcat + ((size_t)(head * 128 + i * 16)) * D_MODEL + kc,
                D_MODEL, lane);
            acc[i] = wmma_bf16(a, b, acc[i]);
        }
    }

    int r0 = lane & 15, h = lane >> 4;
    if (head < 4) {
        unsigned short* outp = (head == 0) ? q1w
                             : (head == 1) ? q2w
                             : (head == 2) ? k1w : k2w;
#pragma unroll
        for (int i = 0; i < 4; ++i) {
            int c = i * 16 + r0;                     // 0..63 (rope pair index)
#pragma unroll
            for (int r = 0; r < 8; ++r) {
                int srow = s0 + r + 8 * h;           // flat row
                int pos  = srow & (SEQ - 1);
                float cs = cosT[(size_t)pos * 64 + c];
                float sn = sinT[(size_t)pos * 64 + c];
                float lo = acc[i][r];                // d = c
                float hi = acc[i + 4][r];            // d = c + 64
                outp[(size_t)srow * D_HEAD + c]      = f2bs(lo * cs - hi * sn);
                outp[(size_t)srow * D_HEAD + 64 + c] = f2bs(lo * sn + hi * cs);
            }
        }
    } else {                                         // v head -> transposed
#pragma unroll
        for (int i = 0; i < 8; ++i) {
            int d = i * 16 + r0;
#pragma unroll
            for (int r = 0; r < 8; ++r) {
                int srow = s0 + r + 8 * h;
                int b = srow >> 12;                  // /4096
                int s = srow & (SEQ - 1);
                Vt[((size_t)b * D_HEAD + d) * SEQ + s] = f2bs(acc[i][r]);
            }
        }
    }
}

// ---------------------------------------------------------------------------
// Kernel 3: bilinear causal attention.  One wave = 16 query rows; stream t in
// chunks of 32.  Score tiles via WMMA, elementwise s1*s2*scale^2 + causal
// mask, bf16 relayout through a private LDS slice, then A*V WMMAs.
// ---------------------------------------------------------------------------
__global__ void __launch_bounds__(128)
attn(const unsigned short* __restrict__ q1w, const unsigned short* __restrict__ q2w,
     const unsigned short* __restrict__ k1w, const unsigned short* __restrict__ k2w,
     const unsigned short* __restrict__ Vt,  unsigned short* __restrict__ Ows) {
    __shared__ unsigned short Sl[4][16 * 32];        // per-wave private slices

    int lane = threadIdx.x & 31;
    int w = threadIdx.x >> 5;
    int gw = blockIdx.x * 4 + w;                     // 1024 waves
    int b  = gw >> 8;                                // batch
    int qb = gw & 255;
    int q0 = qb * 16;                                // within-batch q row base

    // Q tiles (kept resident across the whole t loop)
    v16bf a1[4], a2[4];
#pragma unroll
    for (int kc = 0; kc < 4; ++kc) {
        const size_t qoff = ((size_t)(b * SEQ + q0)) * D_HEAD + kc * 32;
        a1[kc] = load_tile_A(q1w + qoff, D_HEAD, lane);
        a2[kc] = load_tile_A(q2w + qoff, D_HEAD, lane);
    }

    v8f o[8] = {};
    const float scale2 = 1.0f / 128.0f;              // (1/sqrt(128))^2
    int r0 = lane & 15, h = lane >> 4;
    unsigned short* ls = &Sl[w][0];

    int nchunks = (q0 + 15) / 32 + 1;                // causal: t0 <= q_max
#pragma unroll 1
    for (int ci = 0; ci < nchunks; ++ci) {
        int t0 = ci * 32;
        v8f s1a = {}, s1b = {}, s2a = {}, s2b = {};
        const size_t ka = ((size_t)(b * SEQ + t0)) * D_HEAD;
        const size_t kb = ((size_t)(b * SEQ + t0 + 16)) * D_HEAD;
        __builtin_prefetch(k1w + ka + 32 * D_HEAD, 0, 1);
#pragma unroll
        for (int kc = 0; kc < 4; ++kc) {
            v16bf t;
            t = load_tile_B(k1w + ka + kc * 32, D_HEAD, lane);
            s1a = wmma_bf16(a1[kc], t, s1a);
            t = load_tile_B(k1w + kb + kc * 32, D_HEAD, lane);
            s1b = wmma_bf16(a1[kc], t, s1b);
            t = load_tile_B(k2w + ka + kc * 32, D_HEAD, lane);
            s2a = wmma_bf16(a2[kc], t, s2a);
            t = load_tile_B(k2w + kb + kc * 32, D_HEAD, lane);
            s2b = wmma_bf16(a2[kc], t, s2b);
        }
        // elementwise bilinear product + causal mask, write bf16 16x32 tile
#pragma unroll
        for (int r = 0; r < 8; ++r) {
            int srow = q0 + r + 8 * h;               // query index (in batch)
            int ta = t0 + r0, tb = ta + 16;
            float va = (ta <= srow) ? s1a[r] * s2a[r] * scale2 : 0.0f;
            float vb = (tb <= srow) ? s1b[r] * s2b[r] * scale2 : 0.0f;
            ls[(r + 8 * h) * 32 + r0]      = f2bs(va);
            ls[(r + 8 * h) * 32 + 16 + r0] = f2bs(vb);
        }
        asm volatile("s_wait_dscnt 0x0" ::: "memory");  // same-wave LDS RAW
        v16bf at = load_tile_A(ls, 32, lane);
        // O += A(16x32) * V(32x128)
#pragma unroll
        for (int i = 0; i < 8; ++i) {
            v16bf vb_t = load_tile_B(
                Vt + ((size_t)b * D_HEAD + i * 16) * SEQ + t0, SEQ, lane);
            o[i] = wmma_bf16(at, vb_t, o[i]);
        }
    }

    // store O as bf16 row-major (ROWS x 128) for the output projection
#pragma unroll
    for (int i = 0; i < 8; ++i) {
#pragma unroll
        for (int r = 0; r < 8; ++r) {
            size_t row = (size_t)b * SEQ + q0 + r + 8 * h;
            Ows[row * D_HEAD + i * 16 + r0] = f2bs(o[i][r]);
        }
    }
}

// ---------------------------------------------------------------------------
// Kernel 4: output projection  out = O(16384x128) @ Wo(128x1024), f32 out.
// One wave = 16 rows x 64 cols (4 tiles), 16 WMMAs.
// ---------------------------------------------------------------------------
__global__ void __launch_bounds__(128)
outproj(const unsigned short* __restrict__ Ows,
        const unsigned short* __restrict__ Wot,
        float* __restrict__ out) {
    int lane = threadIdx.x & 31;
    int gw = blockIdx.x * 4 + (threadIdx.x >> 5);    // 16384 waves
    int rb = gw >> 4;                                // row block 0..1023
    int cb = gw & 15;                                // 64-col block

    v8f acc[4] = {};
#pragma unroll
    for (int kc = 0; kc < 4; ++kc) {
        v16bf a = load_tile_A(Ows + (size_t)(rb * 16) * D_HEAD + kc * 32,
                              D_HEAD, lane);
#pragma unroll
        for (int i = 0; i < 4; ++i) {
            v16bf bt = load_tile_B(
                Wot + (size_t)(cb * 64 + i * 16) * D_HEAD + kc * 32,
                D_HEAD, lane);
            acc[i] = wmma_bf16(a, bt, acc[i]);
        }
    }
    int r0 = lane & 15, h = lane >> 4;
#pragma unroll
    for (int i = 0; i < 4; ++i) {
#pragma unroll
        for (int r = 0; r < 8; ++r) {
            size_t row = (size_t)(rb * 16 + r + 8 * h);
            out[row * D_MODEL + cb * 64 + i * 16 + r0] = acc[i][r];
        }
    }
}

// ---------------------------------------------------------------------------
// Host launcher.  Workspace layout (bytes, ~26.7 MB total):
//   Wcat 640*1024*2 | Wot 1024*128*2 | q1,q2,k1,k2 4x(16384*128*2)
//   Vt 4*128*4096*2 | O 16384*128*2
// ---------------------------------------------------------------------------
extern "C" void kernel_launch(void* const* d_in, const int* in_sizes, int n_in,
                              void* d_out, int out_size, void* d_ws, size_t ws_size,
                              hipStream_t stream) {
    const float* x    = (const float*)d_in[0];
    const float* cosT = (const float*)d_in[1];
    const float* sinT = (const float*)d_in[2];
    // d_in[3] = causal_mask (recomputed from indices, unused)
    const float* Wq1  = (const float*)d_in[4];
    const float* Wq2  = (const float*)d_in[5];
    const float* Wk1  = (const float*)d_in[6];
    const float* Wk2  = (const float*)d_in[7];
    const float* Wv   = (const float*)d_in[8];
    const float* Wo   = (const float*)d_in[9];
    float* out = (float*)d_out;

    char* ws = (char*)d_ws;
    const size_t SZ_WCAT = (size_t)640 * 1024 * 2;
    const size_t SZ_WOT  = (size_t)1024 * 128 * 2;
    const size_t SZ_QK   = (size_t)ROWS * D_HEAD * 2;
    const size_t SZ_VT   = (size_t)BATCH * D_HEAD * SEQ * 2;

    unsigned short* Wcat = (unsigned short*)(ws);
    unsigned short* Wot  = (unsigned short*)(ws + SZ_WCAT);
    unsigned short* q1w  = (unsigned short*)(ws + SZ_WCAT + SZ_WOT);
    unsigned short* q2w  = (unsigned short*)((char*)q1w + SZ_QK);
    unsigned short* k1w  = (unsigned short*)((char*)q2w + SZ_QK);
    unsigned short* k2w  = (unsigned short*)((char*)k1w + SZ_QK);
    unsigned short* Vt   = (unsigned short*)((char*)k2w + SZ_QK);
    unsigned short* Ows  = (unsigned short*)((char*)Vt + SZ_VT);

    // 1) weight prep: 640*1024 + 1024*128 = 786432 threads
    prep_weights<<<3072, 256, 0, stream>>>(Wq1, Wq2, Wk1, Wk2, Wv, Wo, Wcat, Wot);

    // 2) projections + RoPE: 1024 row-blocks x 5 heads = 5120 waves
    proj_rope<<<1280, 128, 0, stream>>>(x, cosT, sinT, Wcat,
                                        q1w, q2w, k1w, k2w, Vt);

    // 3) attention: 4 batches x 256 q-blocks = 1024 waves
    attn<<<256, 128, 0, stream>>>(q1w, q2w, k1w, k2w, Vt, Ows);

    // 4) output projection: 1024 row-blocks x 16 col-blocks = 16384 waves
    outproj<<<4096, 128, 0, stream>>>(Ows, Wot, out);
}
